// Arthur1_16458314678864
// MI455X (gfx1250) — compile-verified, matrix operand order checked by reading, not run on hardware
//
#include <hip/hip_runtime.h>
#include <hip/hip_bf16.h>

// ---------------------------------------------------------------------------
// GCN (3x GCNConv + BN + ReLU) + 4-layer MLP head for MI455X (gfx1250).
// Dense matmuls use v_wmma_f32_16x16x32_bf16 (wave32 WMMA, f32 accum).
// Edge aggregation uses float global atomics (memory-bound part).
// ---------------------------------------------------------------------------

typedef __attribute__((ext_vector_type(16))) __bf16 v16bf;
typedef __attribute__((ext_vector_type(8)))  float  v8f;

#define N_NODES 50000
#define N_EDGES 800000
#define HC_DIM  96
#define BN_S    0.9999950000374997f   // 1/sqrt(1 + 1e-5)

// ----------------------------- small utility kernels -----------------------

__global__ void k_zero_f32(float* p, long n) {
    long i = (long)blockIdx.x * blockDim.x + threadIdx.x;
    if (i < n) p[i] = 0.0f;
}

__global__ void k_f32_to_bf16(const float* __restrict__ x, __bf16* __restrict__ y, long n) {
    long i = (long)blockIdx.x * blockDim.x + threadIdx.x;
    if (i < n) y[i] = (__bf16)x[i];
}

// Pack a [K x Ncol] f32 weight matrix into per-lane WMMA B fragments (bf16),
// zero-padding columns up to NcolPad. Layout per (ktile, ntile):
// 32 lanes x 16 bf16, matching V_WMMA_F32_16X16X32_BF16 B operand:
//   lane = half*16 + n%16 ; half = which 8-K group ; elements 0..7 -> K=kbase..kbase+7,
//   elements 8..15 -> K=16+kbase..23+kbase (kbase = half*8).
__global__ void k_pack_weight(const float* __restrict__ W, __bf16* __restrict__ P,
                              int K, int Ncol, int NcolPad) {
    int idx = blockIdx.x * blockDim.x + threadIdx.x;
    int total = K * NcolPad;
    if (idx >= total) return;
    int k = idx / NcolPad;
    int n = idx % NcolPad;
    int kt   = k >> 5;
    int kin  = k & 31;
    int nt   = n >> 4;
    int nl   = n & 15;
    int half = (kin >> 3) & 1;
    int ele  = (kin & 7) | (((kin >> 4) & 1) << 3);
    int ntiles = NcolPad >> 4;
    long off = ((((long)kt * ntiles + nt) * 32) + half * 16 + nl) * 16 + ele;
    float v = (n < Ncol) ? W[(long)k * Ncol + n] : 0.0f;
    P[off] = (__bf16)v;
}

// ----------------------------- degree / norm -------------------------------

__global__ void k_deg_init(float* deg, int n) {
    int i = blockIdx.x * blockDim.x + threadIdx.x;
    if (i < n) deg[i] = 1.0f;   // self loop
}

__global__ void k_deg_edges(const int* __restrict__ dst, float* deg, int e) {
    int i = blockIdx.x * blockDim.x + threadIdx.x;
    if (i < e) atomicAdd(&deg[dst[i]], 1.0f);
}

__global__ void k_dinv(const float* __restrict__ deg, float* __restrict__ dinv, int n) {
    int i = blockIdx.x * blockDim.x + threadIdx.x;
    if (i < n) dinv[i] = rsqrtf(deg[i]);
}

// ----------------------------- WMMA GEMM -----------------------------------
// C[M x NcolPad] = A[M x K](bf16, row major) * Bpacked.
// One 16x16 output tile per wave, 8 waves per block. M % 16 == 0, K % 32 == 0.
// Epilogue: optional bias, optional relu; writes f32 (Cf) and/or bf16 (Cbf)
// with row stride `nstore`, masking columns >= nstore (for padded widths).
__global__ void __launch_bounds__(256)
k_gemm_wmma(const __bf16* __restrict__ A, const __bf16* __restrict__ Bp,
            float* __restrict__ Cf, __bf16* __restrict__ Cbf,
            const float* __restrict__ bias,
            int M, int K, int NcolPad, int nstore, int relu) {
    const int wave = threadIdx.x >> 5;
    const int lane = threadIdx.x & 31;
    const int ntiles = NcolPad >> 4;
    const int mtiles = M >> 4;
    const long tile = (long)blockIdx.x * 8 + wave;
    if (tile >= (long)mtiles * ntiles) return;   // wave-uniform: EXEC stays full

    const int mt = (int)(tile / ntiles);
    const int nt = (int)(tile % ntiles);
    const int half  = lane >> 4;          // 0: lanes 0-15, 1: lanes 16-31
    const int ml    = lane & 15;
    const int arow  = mt * 16 + ml;       // A row held by this lane
    const int kbase = half * 8;

    const __bf16* __restrict__ ap = A + (long)arow * K;
    const v16bf*  __restrict__ bp = (const v16bf*)Bp;

    v8f acc = {};
    const int ksteps = K >> 5;
    for (int kt = 0; kt < ksteps; ++kt) {
        union { uint4 u[2]; v16bf v; } ua;
        ua.u[0] = *(const uint4*)(ap + kt * 32 + kbase);        // K = kbase..kbase+7
        ua.u[1] = *(const uint4*)(ap + kt * 32 + 16 + kbase);   // K = 16+kbase..23+kbase
        v16bf bv = bp[((long)kt * ntiles + nt) * 32 + lane];
        acc = __builtin_amdgcn_wmma_f32_16x16x32_bf16(
                  false, ua.v, false, bv, (short)0, acc, false, false);
    }

    const int n = nt * 16 + ml;           // output column for this lane
    float bb = 0.0f;
    if (bias != nullptr && n < nstore) bb = bias[n];

    if (n < nstore) {
        #pragma unroll
        for (int r = 0; r < 8; ++r) {
            int m = mt * 16 + half * 8 + r;
            float v = acc[r] + bb;
            if (relu) v = fmaxf(v, 0.0f);
            if (Cf)  Cf [(long)m * nstore + n] = v;
            if (Cbf) Cbf[(long)m * nstore + n] = (__bf16)v;
        }
    }
}

// ----------------------------- edge scatter --------------------------------
// AGG[dst] += dinv[src]*dinv[dst] * H[src]; one thread per (edge, 4 features).
__global__ void k_scatter(const float* __restrict__ H,
                          const int* __restrict__ src, const int* __restrict__ dst,
                          const float* __restrict__ dinv, float* __restrict__ AGG,
                          int E) {
    long t = (long)blockIdx.x * blockDim.x + threadIdx.x;
    long total = (long)E * (HC_DIM / 4);
    if (t >= total) return;
    int e = (int)(t / (HC_DIM / 4));
    int c = (int)(t % (HC_DIM / 4)) * 4;
    int s = src[e], d = dst[e];
    float nrm = dinv[s] * dinv[d];
    float4 h = *(const float4*)(H + (long)s * HC_DIM + c);
    float* out = AGG + (long)d * HC_DIM + c;
    atomicAdd(out + 0, nrm * h.x);
    atomicAdd(out + 1, nrm * h.y);
    atomicAdd(out + 2, nrm * h.z);
    atomicAdd(out + 3, nrm * h.w);
}

// ----------------------------- GCN epilogue --------------------------------
// val = relu( (AGG + dinv^2 * H (self loop) + b) * BN_S * g + be ) -> bf16 act
__global__ void k_gcn_epilogue(const float* __restrict__ AGG, const float* __restrict__ H,
                               const float* __restrict__ dinv,
                               const float* __restrict__ b, const float* __restrict__ g,
                               const float* __restrict__ be,
                               __bf16* __restrict__ act, long n) {
    long i = (long)blockIdx.x * blockDim.x + threadIdx.x;
    if (i >= n) return;
    int node = (int)(i / HC_DIM);
    int c    = (int)(i % HC_DIM);
    float di = dinv[node];
    float v = AGG[i] + di * di * H[i] + b[c];
    v = v * BN_S * g[c] + be[c];
    v = fmaxf(v, 0.0f);
    act[i] = (__bf16)v;
}

// ----------------------------- host side -----------------------------------

static inline size_t align256(size_t x) { return (x + 255) & ~(size_t)255; }

extern "C" void kernel_launch(void* const* d_in, const int* in_sizes, int n_in,
                              void* d_out, int out_size, void* d_ws, size_t ws_size,
                              hipStream_t stream) {
    (void)in_sizes; (void)n_in; (void)out_size; (void)ws_size;

    const float* x   = (const float*)d_in[0];
    const int*   ei  = (const int*)d_in[1];          // (2, E): row0=src, row1=dst
    const float* w1  = (const float*)d_in[2];
    const float* b1  = (const float*)d_in[3];
    const float* g1  = (const float*)d_in[4];
    const float* be1 = (const float*)d_in[5];
    const float* w2  = (const float*)d_in[6];
    const float* b2  = (const float*)d_in[7];
    const float* g2  = (const float*)d_in[8];
    const float* be2 = (const float*)d_in[9];
    const float* w3  = (const float*)d_in[10];
    const float* b3  = (const float*)d_in[11];
    const float* g3  = (const float*)d_in[12];
    const float* be3 = (const float*)d_in[13];
    const float* lw1 = (const float*)d_in[14];
    const float* lb1 = (const float*)d_in[15];
    const float* lw2 = (const float*)d_in[16];
    const float* lb2 = (const float*)d_in[17];
    const float* lw3 = (const float*)d_in[18];
    const float* lb3 = (const float*)d_in[19];
    const float* lw4 = (const float*)d_in[20];
    const float* lb4 = (const float*)d_in[21];

    const int N = N_NODES, E = N_EDGES;
    const int* src = ei;
    const int* dst = ei + E;

    // ---- workspace layout ----
    char* w = (char*)d_ws;
    float* deg  = (float*)w;  w += align256((size_t)N * 4);
    float* dinv = (float*)w;  w += align256((size_t)N * 4);
    float* H    = (float*)w;  w += align256((size_t)N * HC_DIM * 4);
    float* AGG  = (float*)w;  w += align256((size_t)N * HC_DIM * 4);
    __bf16* actA = (__bf16*)w; w += align256((size_t)N * 256 * 2);
    __bf16* actB = (__bf16*)w; w += align256((size_t)N * 256 * 2);
    __bf16* w1p  = (__bf16*)w; w += align256((size_t)64  * 96  * 2);
    __bf16* w2p  = (__bf16*)w; w += align256((size_t)96  * 96  * 2);
    __bf16* w3p  = (__bf16*)w; w += align256((size_t)96  * 96  * 2);
    __bf16* lw1p = (__bf16*)w; w += align256((size_t)96  * 256 * 2);
    __bf16* lw2p = (__bf16*)w; w += align256((size_t)256 * 128 * 2);
    __bf16* lw3p = (__bf16*)w; w += align256((size_t)128 * 64  * 2);
    __bf16* lw4p = (__bf16*)w; w += align256((size_t)64  * 16  * 2);

    const int T = 256;
    auto blocks = [](long n, int t) { return (unsigned)((n + t - 1) / t); };

    // ---- pack weights to WMMA B fragments (bf16) ----
    k_pack_weight<<<blocks(64  * 96 , T), T, 0, stream>>>(w1 , w1p , 64 , 96 , 96 );
    k_pack_weight<<<blocks(96  * 96 , T), T, 0, stream>>>(w2 , w2p , 96 , 96 , 96 );
    k_pack_weight<<<blocks(96  * 96 , T), T, 0, stream>>>(w3 , w3p , 96 , 96 , 96 );
    k_pack_weight<<<blocks(96  * 256, T), T, 0, stream>>>(lw1, lw1p, 96 , 256, 256);
    k_pack_weight<<<blocks(256 * 128, T), T, 0, stream>>>(lw2, lw2p, 256, 128, 128);
    k_pack_weight<<<blocks(128 * 64 , T), T, 0, stream>>>(lw3, lw3p, 128, 64 , 64 );
    k_pack_weight<<<blocks(64  * 16 , T), T, 0, stream>>>(lw4, lw4p, 64 , 8  , 16 );

    // ---- x -> bf16 activations ----
    k_f32_to_bf16<<<blocks((long)N * 64, T), T, 0, stream>>>(x, actA, (long)N * 64);

    // ---- degrees / normalization ----
    k_deg_init <<<blocks(N, T), T, 0, stream>>>(deg, N);
    k_deg_edges<<<blocks(E, T), T, 0, stream>>>(dst, deg, E);
    k_dinv     <<<blocks(N, T), T, 0, stream>>>(deg, dinv, N);

    const long nh = (long)N * HC_DIM;
    const unsigned scatterB = blocks((long)E * (HC_DIM / 4), T);

    // ---- GCN layer 1: H = xbf @ w1 ; aggregate ; epilogue -> actB ----
    {
        long tiles = ((long)N / 16) * (96 / 16);
        k_gemm_wmma<<<blocks(tiles * 32, T), T, 0, stream>>>(
            actA, w1p, H, nullptr, nullptr, N, 64, 96, 96, 0);
        k_zero_f32<<<blocks(nh, T), T, 0, stream>>>(AGG, nh);
        k_scatter<<<scatterB, T, 0, stream>>>(H, src, dst, dinv, AGG, E);
        k_gcn_epilogue<<<blocks(nh, T), T, 0, stream>>>(AGG, H, dinv, b1, g1, be1, actB, nh);
    }
    // ---- GCN layer 2 ----
    {
        long tiles = ((long)N / 16) * (96 / 16);
        k_gemm_wmma<<<blocks(tiles * 32, T), T, 0, stream>>>(
            actB, w2p, H, nullptr, nullptr, N, 96, 96, 96, 0);
        k_zero_f32<<<blocks(nh, T), T, 0, stream>>>(AGG, nh);
        k_scatter<<<scatterB, T, 0, stream>>>(H, src, dst, dinv, AGG, E);
        k_gcn_epilogue<<<blocks(nh, T), T, 0, stream>>>(AGG, H, dinv, b2, g2, be2, actA, nh);
    }
    // ---- GCN layer 3 ----
    {
        long tiles = ((long)N / 16) * (96 / 16);
        k_gemm_wmma<<<blocks(tiles * 32, T), T, 0, stream>>>(
            actA, w3p, H, nullptr, nullptr, N, 96, 96, 96, 0);
        k_zero_f32<<<blocks(nh, T), T, 0, stream>>>(AGG, nh);
        k_scatter<<<scatterB, T, 0, stream>>>(H, src, dst, dinv, AGG, E);
        k_gcn_epilogue<<<blocks(nh, T), T, 0, stream>>>(AGG, H, dinv, b3, g3, be3, actB, nh);
    }

    // ---- MLP head (bias+relu fused in GEMM epilogue, bf16 ping-pong) ----
    {
        long tiles = ((long)N / 16) * (256 / 16);
        k_gemm_wmma<<<blocks(tiles * 32, T), T, 0, stream>>>(
            actB, lw1p, nullptr, actA, lb1, N, 96, 256, 256, 1);
    }
    {
        long tiles = ((long)N / 16) * (128 / 16);
        k_gemm_wmma<<<blocks(tiles * 32, T), T, 0, stream>>>(
            actA, lw2p, nullptr, actB, lb2, N, 256, 128, 128, 1);
    }
    {
        long tiles = ((long)N / 16) * (64 / 16);
        k_gemm_wmma<<<blocks(tiles * 32, T), T, 0, stream>>>(
            actB, lw3p, nullptr, actA, lb3, N, 128, 64, 64, 1);
    }
    {   // final 64 -> 8 (padded to 16 cols, store only 8) straight to d_out (f32)
        long tiles = ((long)N / 16) * (16 / 16);
        k_gemm_wmma<<<blocks(tiles * 32, T), T, 0, stream>>>(
            actA, lw4p, (float*)d_out, nullptr, lb4, N, 64, 16, 8, 0);
    }
}
